// SLR_Net_55405078118625
// MI455X (gfx1250) — matrix operand order repaired and checked by reading, not compiled, not run on hardware
//
#include <hip/hip_runtime.h>
#include <hip/hip_bf16.h>

// ---------------------------------------------------------------------------
// SLR-Net for MI455X (gfx1250, wave32, WMMA, async-to-LDS)
//   * convs (dominant ~150 GFLOP) -> v_wmma_f32_16x16x32_f16, f32 accumulate,
//     weights staged in LDS via global_load_async_to_lds_b128
//   * fft2c/ifft2c -> symmetric shifted-DFT matrix F; each pass computes
//     (A @ F)^T with v_wmma_f32_16x16x4_f32; two passes = F X F = 2D transform.
//     Both Fr and Fi (200KB) live in LDS (CDNA5: 320KB/WGP).
//   * SVD -> Gram matrix + real Jacobi on 36x36 embedding (VALU, negligible)
// ---------------------------------------------------------------------------

typedef __attribute__((ext_vector_type(16))) _Float16 v16h;
typedef __attribute__((ext_vector_type(8)))  float    v8f;
typedef __attribute__((ext_vector_type(2)))  float    v2f;

#define NT    18
#define NX    160
#define NY    160
#define NPIX  (NX * NY)              // 25600
#define NTPIX (NT * NPIX)            // 460800
#define NITER 3
#define PACKH (14 * 32 * 16)         // halfs per (layer,iter) weight pack = 7168

// --------------------- CDNA5 async global->LDS staging ---------------------
// 16B per lane; tracked by ASYNCcnt (ISA ch10.4 / 15.18.3 opcode 98).
__device__ __forceinline__ void async_cp16(void* lds_dst, const void* gsrc) {
  unsigned           ldso = (unsigned)(unsigned long long)lds_dst; // low 32b = LDS offset
  unsigned long long ga   = (unsigned long long)gsrc;
  asm volatile("global_load_async_to_lds_b128 %0, %1, off"
               :: "v"(ldso), "v"(ga) : "memory");
}
__device__ __forceinline__ void async_wait_all() {
  asm volatile("s_wait_asynccnt 0x0" ::: "memory");
}

// ------------------------------- prep kernels ------------------------------

// Shifted orthonormal DFT matrix: F[j][k] = exp(-2*pi*i*((j+80)%160)*((k+80)%160)/160)/sqrt(160)
// Symmetric in (j,k); Fi stored as the imaginary part of the *forward* transform.
__global__ void k_buildF(float* __restrict__ Fr, float* __restrict__ Fi) {
  int idx = blockIdx.x * blockDim.x + threadIdx.x;
  if (idx >= NPIX) return;
  int j = idx / NY, k = idx % NY;
  int jj = (j + 80) % 160, kk = (k + 80) % 160;
  int ph = (jj * kk) % 160;                       // periodic phase, keeps angle small
  float ang = 6.28318530717958647f * (float)ph / 160.f;
  const float inv = 0.0790569415042094833f;       // 1/sqrt(160)
  Fr[idx] = cosf(ang) * inv;
  Fi[idx] = -sinf(ang) * inv;
}

// Pack conv weights [NITER][3][3][3][Cin][Cout] into WMMA-B order:
// pack[it][chunk][lane][e]; lane: n=lane&15, k=(e>>3)*16+(lane>>4)*8+(e&7); tap=2c+(k>>4); ci=k&15
__global__ void k_pack(const float* __restrict__ W, _Float16* __restrict__ out,
                       int Cin, int Cout) {
  int idx = blockIdx.x * blockDim.x + threadIdx.x;
  if (idx >= NITER * PACKH) return;
  int e    = idx & 15;
  int lane = (idx >> 4) & 31;
  int c    = (idx >> 9) % 14;
  int it   = idx / PACKH;
  int n  = lane & 15;
  int k  = ((e >> 3) << 4) + ((lane >> 4) << 3) + (e & 7);
  int tap = 2 * c + (k >> 4);
  int ci  = k & 15;
  float v = 0.f;
  if (tap < 27 && ci < Cin && n < Cout)
    v = W[(((size_t)it * 27 + tap) * Cin + ci) * Cout + n];
  out[idx] = (_Float16)v;
}

__global__ void k_zero(float* __restrict__ p, int n) {
  int idx = blockIdx.x * blockDim.x + threadIdx.x;
  if (idx < n) p[idx] = 0.f;
}

// --------------------- complex GEMM-transpose (DFT pass) -------------------
// out = (A @ (Fr + i*fsign*Fi))^T per 18 slices; F symmetric so two passes
// give the full 2D (i)fft2c. B operands read from LDS (200KB staged async).
__global__ __launch_bounds__(128) void k_cgemmT(
    float* __restrict__ Cr, float* __restrict__ Ci,
    const float* __restrict__ Ar, const float* __restrict__ Ai,
    const float* __restrict__ Fr, const float* __restrict__ Fi, float fsign) {
  extern __shared__ float sF[];        // [0,25600): Fr  [25600,51200): Fi
  int tid = threadIdx.x;
  for (int k = tid; k < 6400; k += 128) {          // 2 x 100KB, 16B per op
    async_cp16(sF + (size_t)k * 4,         Fr + (size_t)k * 4);
    async_cp16(sF + 25600 + (size_t)k * 4, Fi + (size_t)k * 4);
  }
  async_wait_all();
  __syncthreads();

  int wave = blockIdx.x * 4 + (tid >> 5);          // 0..1799
  int b   = wave / 100;
  int mt  = (wave % 100) / 10;
  int ntl = wave % 10;
  const float* Abr = Ar + (size_t)b * NPIX;
  const float* Abi = Ai + (size_t)b * NPIX;
  float*       Cbr = Cr + (size_t)b * NPIX;
  float*       Cbi = Ci + (size_t)b * NPIX;
  int l = tid & 31, m = l & 15, hi = l >> 4;
  int row = mt * 16 + m;             // A row (M)
  int col = ntl * 16 + m;            // F row == B column (F symmetric)
  v8f aRR = {0.f,0.f,0.f,0.f,0.f,0.f,0.f,0.f};
  v8f aRI = aRR, aIR = aRR, aII = aRR;
#pragma unroll 2
  for (int kk = 0; kk < 160; kk += 4) {
    v2f ar = *(const v2f*)(Abr + row * 160 + kk + 2 * hi);
    v2f ai = *(const v2f*)(Abi + row * 160 + kk + 2 * hi);
    v2f br = *(const v2f*)(sF + col * 160 + kk + 2 * hi);           // ds_load
    v2f bi = *(const v2f*)(sF + 25600 + col * 160 + kk + 2 * hi);   // ds_load
    aRR = __builtin_amdgcn_wmma_f32_16x16x4_f32(false, ar, false, br, (short)0, aRR, false, false);
    aRI = __builtin_amdgcn_wmma_f32_16x16x4_f32(false, ai, false, bi, (short)0, aRI, false, false);
    aIR = __builtin_amdgcn_wmma_f32_16x16x4_f32(false, ar, false, bi, (short)0, aIR, false, false);
    aII = __builtin_amdgcn_wmma_f32_16x16x4_f32(false, ai, false, br, (short)0, aII, false, false);
  }
  // transposed store: output row = col index n, columns mt*16+8*hi..+7 (contiguous)
  float s = fsign;
  float orv[8], oiv[8];
#pragma unroll
  for (int v = 0; v < 8; ++v) {
    orv[v] = aRR[v] - s * aRI[v];
    oiv[v] = s * aIR[v] + aII[v];
  }
  float* cpr = Cbr + (size_t)(ntl * 16 + m) * 160 + mt * 16 + 8 * hi;
  float* cpi = Cbi + (size_t)(ntl * 16 + m) * 160 + mt * 16 + 8 * hi;
  *(float4*)(cpr + 0) = make_float4(orv[0], orv[1], orv[2], orv[3]);
  *(float4*)(cpr + 4) = make_float4(orv[4], orv[5], orv[6], orv[7]);
  *(float4*)(cpi + 0) = make_float4(oiv[0], oiv[1], oiv[2], oiv[3]);
  *(float4*)(cpi + 4) = make_float4(oiv[4], oiv[5], oiv[6], oiv[7]);
}

// ------------------------------- f16 WMMA conv -----------------------------
// 3x3x3 SAME conv over [NT,NX,NY] volume, 16-ch padded activations.
// One wave per (t, x, 16-wide y tile); 14 K=32 WMMAs (2 taps per chunk).
// Weight pack (14KB) staged once per block into LDS, shared by all 4 waves.
__global__ __launch_bounds__(128) void k_conv(
    const _Float16* __restrict__ in, const _Float16* __restrict__ wpack,
    _Float16* __restrict__ out, int relu_flag) {
  __shared__ _Float16 sw[PACKH];     // 14336 bytes
  int tid = threadIdx.x;
  for (int k = tid; k < PACKH / 8; k += 128)       // 896 x 16B
    async_cp16(sw + (size_t)k * 8, wpack + (size_t)k * 8);
  async_wait_all();
  __syncthreads();

  int wave = blockIdx.x * 4 + (tid >> 5);          // 0..28799
  int t   = wave / 1600;
  int rem = wave % 1600;
  int x   = rem / 10;
  int y0  = (rem % 10) * 16;
  int l = tid & 31, m = l & 15, hi = l >> 4;
  v8f acc = {0.f,0.f,0.f,0.f,0.f,0.f,0.f,0.f};
  for (int c = 0; c < 14; ++c) {
    union { int4 q[2]; v16h v; } a;
#pragma unroll
    for (int j = 0; j < 2; ++j) {
      int tap = 2 * c + j;
      int4 val = make_int4(0, 0, 0, 0);
      if (tap < 27) {
        int dt = tap / 9 - 1;
        int r9 = tap % 9;
        int dx = r9 / 3 - 1, dy = r9 % 3 - 1;
        int tt = t + dt, xx = x + dx, yy = y0 + m + dy;
        if ((unsigned)tt < (unsigned)NT && (unsigned)xx < (unsigned)NX &&
            (unsigned)yy < (unsigned)NY) {
          // elements j*8..j*8+7  <->  K = j*16 + 8*hi + 0..7  (ci = 8*hi+0..7)
          val = *(const int4*)(in + ((((size_t)tt * NX + xx) * NY + yy) << 4) + (hi << 3));
        }
      }
      a.q[j] = val;
    }
    v16h b = *(const v16h*)(sw + ((size_t)c * 32 + l) * 16);   // ds_load_b128 x2
    acc = __builtin_amdgcn_wmma_f32_16x16x32_f16(false, a.v, false, b,
                                                 (short)0, acc, false, false);
  }
  // D: lane -> col n=m, rows M = 8*hi + v
  _Float16* op = out + (((size_t)t * NX + x) * NY + y0 + 8 * hi) * 16 + m;
#pragma unroll
  for (int v = 0; v < 8; ++v) {
    float val = acc[v];
    if (relu_flag) val = val > 0.f ? val : 0.f;
    op[(size_t)v * 16] = (_Float16)val;
  }
}

// ------------------------------- elementwise -------------------------------

__global__ void k_premask(const float* __restrict__ dr, const float* __restrict__ di,
                          const int* __restrict__ msk,
                          float* __restrict__ oR, float* __restrict__ oI) {
  int idx = blockIdx.x * blockDim.x + threadIdx.x;
  if (idx >= NTPIX) return;
  float mv = (float)msk[idx];
  oR[idx] = dr[idx] * mv;
  oI[idx] = di[idx] * mv;
}

__global__ void k_dc(const float* __restrict__ zR, const float* __restrict__ zI,
                     const float* __restrict__ dr, const float* __restrict__ di,
                     const int* __restrict__ msk,
                     float* __restrict__ oR, float* __restrict__ oI) {
  int idx = blockIdx.x * blockDim.x + threadIdx.x;
  if (idx >= NTPIX) return;
  float mv = (float)msk[idx];
  oR[idx] = mv * (mv * zR[idx] - dr[idx]);
  oI[idx] = mv * (mv * zI[idx] - di[idx]);
}

__global__ void k_rn(const float* __restrict__ xR, const float* __restrict__ xI,
                     const float* __restrict__ AtR, const float* __restrict__ AtI,
                     const float* __restrict__ tR, const float* __restrict__ tI,
                     const float* __restrict__ bR, const float* __restrict__ bI,
                     const float* __restrict__ lam1, const float* __restrict__ lam2,
                     int iter, float* __restrict__ rR, float* __restrict__ rI,
                     _Float16* __restrict__ actR) {
  int idx = blockIdx.x * blockDim.x + threadIdx.x;
  if (idx >= NTPIX) return;
  float l1 = fmaxf(lam1[iter], 0.f), l2 = fmaxf(lam2[iter], 0.f);
  float rr = xR[idx] - l1 * AtR[idx] + l2 * (xR[idx] + bR[idx] - tR[idx]);
  float ri = xI[idx] - l1 * AtI[idx] + l2 * (xI[idx] + bI[idx] - tI[idx]);
  rR[idx] = rr;
  rI[idx] = ri;
  union { _Float16 h[16]; int4 q[2]; } u;
  u.q[0] = make_int4(0, 0, 0, 0);
  u.q[1] = make_int4(0, 0, 0, 0);
  u.h[0] = (_Float16)rr;
  u.h[1] = (_Float16)ri;
  int4* o = (int4*)(actR + (size_t)idx * 16);
  o[0] = u.q[0];
  o[1] = u.q[1];
}

__global__ void k_soft(const _Float16* __restrict__ in, _Float16* __restrict__ out,
                       const float* __restrict__ thr_p, int iter) {
  int idx = blockIdx.x * blockDim.x + threadIdx.x;
  if (idx >= NTPIX * 16) return;
  float v = (float)in[idx];
  float thr = thr_p[iter];
  float mag = fmaxf(fabsf(v) - thr, 0.f);
  float s = (v > 0.f) ? 1.f : ((v < 0.f) ? -1.f : 0.f);
  out[idx] = (_Float16)(s * mag);
}

__global__ void k_combine(const _Float16* __restrict__ act,
                          const float* __restrict__ rR, const float* __restrict__ rI,
                          float* __restrict__ xR, float* __restrict__ xI) {
  int idx = blockIdx.x * blockDim.x + threadIdx.x;
  if (idx >= NTPIX) return;
  xR[idx] = rR[idx] + (float)act[(size_t)idx * 16 + 0];
  xI[idx] = rI[idx] + (float)act[(size_t)idx * 16 + 1];
}

__global__ void k_xsym(const _Float16* __restrict__ act,
                       const float* __restrict__ rR, const float* __restrict__ rI,
                       float* __restrict__ out) {
  int idx = blockIdx.x * blockDim.x + threadIdx.x;
  if (idx >= NTPIX) return;
  out[(size_t)idx * 2 + 0] = (float)act[(size_t)idx * 16 + 0] - rR[idx];
  out[(size_t)idx * 2 + 1] = (float)act[(size_t)idx * 16 + 1] - rI[idx];
}

__global__ void k_writeout(const float* __restrict__ xR, const float* __restrict__ xI,
                           float* __restrict__ out) {
  int idx = blockIdx.x * blockDim.x + threadIdx.x;
  if (idx >= NTPIX) return;
  out[(size_t)idx * 2 + 0] = xR[idx];
  out[(size_t)idx * 2 + 1] = xI[idx];
}

// ------------------------------- low-rank step -----------------------------

// G = M M^H, one block per (a,b) entry
__global__ void k_gram(const float* __restrict__ xR, const float* __restrict__ xI,
                       float* __restrict__ Gr, float* __restrict__ Gi) {
  __shared__ float sr[256], si[256];
  int a = blockIdx.x / NT, b = blockIdx.x % NT;
  const float* ar = xR + (size_t)a * NPIX;
  const float* ai = xI + (size_t)a * NPIX;
  const float* br = xR + (size_t)b * NPIX;
  const float* bi = xI + (size_t)b * NPIX;
  float accr = 0.f, acci = 0.f;
  for (int p = threadIdx.x; p < NPIX; p += 256) {
    float xar = ar[p], xai = ai[p], xbr = br[p], xbi = bi[p];
    accr += xar * xbr + xai * xbi;
    acci += xai * xbr - xar * xbi;
  }
  sr[threadIdx.x] = accr;
  si[threadIdx.x] = acci;
  __syncthreads();
  for (int s = 128; s > 0; s >>= 1) {
    if (threadIdx.x < s) {
      sr[threadIdx.x] += sr[threadIdx.x + s];
      si[threadIdx.x] += si[threadIdx.x + s];
    }
    __syncthreads();
  }
  if (threadIdx.x == 0) { Gr[blockIdx.x] = sr[0]; Gi[blockIdx.x] = si[0]; }
}

// Jacobi eig of real embedding [[Gr,-Gi],[Gi,Gr]]; C = Q diag(relu(s-thr)/s) Q^T
__global__ void k_eig(const float* __restrict__ Gr, const float* __restrict__ Gi,
                      const float* __restrict__ thres_coef, int iter,
                      float* __restrict__ Cr, float* __restrict__ Ci) {
  __shared__ float A[36][36];
  __shared__ float Q[36][36];
  __shared__ float coef[36];
  int tid = threadIdx.x;
  for (int e = tid; e < 36 * 36; e += blockDim.x) {
    int r = e / 36, c = e % 36;
    int a = r % 18, b = c % 18;
    float v;
    if (r < 18 && c < 18) v = Gr[a * 18 + b];
    else if (r < 18)      v = -Gi[a * 18 + b];
    else if (c < 18)      v = Gi[a * 18 + b];
    else                  v = Gr[a * 18 + b];
    A[r][c] = v;
    Q[r][c] = (r == c) ? 1.f : 0.f;
  }
  __syncthreads();
  if (tid == 0) {
    for (int sweep = 0; sweep < 16; ++sweep) {
      for (int p = 0; p < 35; ++p)
        for (int q = p + 1; q < 36; ++q) {
          float apq = A[p][q];
          if (fabsf(apq) < 1e-12f) continue;
          float th = 0.5f * atan2f(2.f * apq, A[q][q] - A[p][p]);
          float c = cosf(th), s = sinf(th);
          for (int k = 0; k < 36; ++k) {           // A <- A J
            float akp = A[k][p], akq = A[k][q];
            A[k][p] = c * akp - s * akq;
            A[k][q] = s * akp + c * akq;
          }
          for (int k = 0; k < 36; ++k) {           // A <- J^T A ; Q <- Q J
            float apk = A[p][k], aqk = A[q][k];
            A[p][k] = c * apk - s * aqk;
            A[q][k] = s * apk + c * aqk;
            float qkp = Q[k][p], qkq = Q[k][q];
            Q[k][p] = c * qkp - s * qkq;
            Q[k][q] = s * qkp + c * qkq;
          }
        }
    }
    float smax = 0.f;
    for (int j = 0; j < 36; ++j) {
      float lam = A[j][j];
      float s = lam > 0.f ? sqrtf(lam) : 0.f;
      coef[j] = s;
      smax = fmaxf(smax, s);
    }
    float sig = 1.f / (1.f + expf(-thres_coef[iter]));
    float thr = sig * smax;
    for (int j = 0; j < 36; ++j) {
      float s = coef[j];
      coef[j] = (s > 1e-20f) ? fmaxf(s - thr, 0.f) / s : 0.f;
    }
  }
  __syncthreads();
  for (int e = tid; e < 324; e += blockDim.x) {
    int a = e / 18, b = e % 18;
    float cr = 0.f, ci = 0.f;
    for (int j = 0; j < 36; ++j) {
      float w = coef[j] * Q[b][j];
      cr += w * Q[a][j];
      ci += w * Q[18 + a][j];
    }
    Cr[e] = cr;
    Ci[e] = ci;
  }
}

// t = C @ M (complex 18x18 per pixel); beta += eta*(x - t)
__global__ void k_lowrank(const float* __restrict__ xR, const float* __restrict__ xI,
                          const float* __restrict__ Cr, const float* __restrict__ Ci,
                          float* __restrict__ tR, float* __restrict__ tI,
                          float* __restrict__ bR, float* __restrict__ bI,
                          const float* __restrict__ eta_p, int iter) {
  __shared__ float sCr[324], sCi[324];
  for (int e = threadIdx.x; e < 324; e += blockDim.x) { sCr[e] = Cr[e]; sCi[e] = Ci[e]; }
  __syncthreads();
  int p = blockIdx.x * blockDim.x + threadIdx.x;
  if (p >= NPIX) return;
  float eta = fmaxf(eta_p[iter], 0.f);
  float xr[18], xi[18];
#pragma unroll
  for (int b = 0; b < 18; ++b) {
    xr[b] = xR[(size_t)b * NPIX + p];
    xi[b] = xI[(size_t)b * NPIX + p];
  }
  for (int a = 0; a < 18; ++a) {
    float tr = 0.f, ti = 0.f;
#pragma unroll
    for (int b = 0; b < 18; ++b) {
      float cr = sCr[a * 18 + b], ci = sCi[a * 18 + b];
      tr += cr * xr[b] - ci * xi[b];
      ti += cr * xi[b] + ci * xr[b];
    }
    size_t o = (size_t)a * NPIX + p;
    tR[o] = tr;
    tI[o] = ti;
    bR[o] += eta * (xr[a] - tr);
    bI[o] += eta * (xi[a] - ti);
  }
}

// ------------------------------- host helpers ------------------------------

// out = (i)fft2c(in): two (A@F)^T passes (F symmetric -> F X F)
static void fft_apply(hipStream_t s, float* outR, float* outI,
                      const float* inR, const float* inI,
                      float* tmpR, float* tmpI,
                      const float* Fr, const float* Fi, float fsign) {
  k_cgemmT<<<dim3(450), dim3(128), 204800, s>>>(tmpR, tmpI, inR, inI, Fr, Fi, fsign);
  k_cgemmT<<<dim3(450), dim3(128), 204800, s>>>(outR, outI, tmpR, tmpI, Fr, Fi, fsign);
}

extern "C" void kernel_launch(void* const* d_in, const int* in_sizes, int n_in,
                              void* d_out, int out_size, void* d_ws, size_t ws_size,
                              hipStream_t stream) {
  (void)in_sizes; (void)n_in; (void)out_size; (void)ws_size;
  const float* d_real = (const float*)d_in[0];
  const float* d_imag = (const float*)d_in[1];
  const int*   mask   = (const int*)d_in[2];
  const float* Wl[6];
  for (int i = 0; i < 6; ++i) Wl[i] = (const float*)d_in[3 + i];
  const float* lam1       = (const float*)d_in[9];
  const float* lam2       = (const float*)d_in[10];
  const float* soft_thr   = (const float*)d_in[11];
  const float* thres_coef = (const float*)d_in[12];
  const float* eta        = (const float*)d_in[13];
  float* out = (float*)d_out;

  // bump allocator over d_ws
  char* wptr = (char*)d_ws;
  auto alloc = [&](size_t bytes) -> char* {
    char* p = wptr;
    wptr += (bytes + 255) & ~(size_t)255;
    return p;
  };
  const size_t PB = (size_t)NTPIX * 4;             // plane bytes
  float* Fr  = (float*)alloc(NPIX * 4);
  float* Fi  = (float*)alloc(NPIX * 4);
  float* xR  = (float*)alloc(PB);  float* xI  = (float*)alloc(PB);
  float* tR  = (float*)alloc(PB);  float* tI  = (float*)alloc(PB);
  float* bR  = (float*)alloc(PB);  float* bI  = (float*)alloc(PB);
  float* t1R = (float*)alloc(PB);  float* t1I = (float*)alloc(PB);
  float* t2R = (float*)alloc(PB);  float* t2I = (float*)alloc(PB);
  float* t3R = (float*)alloc(PB);  float* t3I = (float*)alloc(PB);
  float* rR  = (float*)alloc(PB);  float* rI  = (float*)alloc(PB);
  const size_t AB = (size_t)NTPIX * 16 * 2;        // f16 activation bytes
  _Float16* actR  = (_Float16*)alloc(AB);
  _Float16* actA  = (_Float16*)alloc(AB);
  _Float16* actB  = (_Float16*)alloc(AB);
  _Float16* actX3 = (_Float16*)alloc(AB);
  _Float16* packW = (_Float16*)alloc((size_t)6 * NITER * PACKH * 2);
  float* Gr = (float*)alloc(324 * 4);  float* Gi = (float*)alloc(324 * 4);
  float* Cr = (float*)alloc(324 * 4);  float* Ci = (float*)alloc(324 * 4);

  const dim3 EW_G(1800), EW_B(256);                // 460800 element kernels

  // ---- prep ----
  k_buildF<<<dim3(100), dim3(256), 0, stream>>>(Fr, Fi);
  const int CinA[6]  = {2, 16, 16, 16, 16, 16};
  const int CoutA[6] = {16, 16, 16, 16, 16, 2};
  for (int l = 0; l < 6; ++l)
    k_pack<<<dim3(84), dim3(256), 0, stream>>>(Wl[l], packW + (size_t)l * NITER * PACKH,
                                               CinA[l], CoutA[l]);
  k_zero<<<EW_G, EW_B, 0, stream>>>(tR, NTPIX);
  k_zero<<<EW_G, EW_B, 0, stream>>>(tI, NTPIX);
  k_zero<<<EW_G, EW_B, 0, stream>>>(bR, NTPIX);
  k_zero<<<EW_G, EW_B, 0, stream>>>(bI, NTPIX);

  // x_rec = ifft2c(d * m)
  k_premask<<<EW_G, EW_B, 0, stream>>>(d_real, d_imag, mask, t1R, t1I);
  fft_apply(stream, xR, xI, t1R, t1I, t2R, t2I, Fr, Fi, -1.f);

  for (int i = 0; i < NITER; ++i) {
    // ATAX = ifft2c(m*(m*fft2c(x) - d))
    fft_apply(stream, t2R, t2I, xR, xI, t1R, t1I, Fr, Fi, 1.f);
    k_dc<<<EW_G, EW_B, 0, stream>>>(t2R, t2I, d_real, d_imag, mask, t1R, t1I);
    fft_apply(stream, t2R, t2I, t1R, t1I, t3R, t3I, Fr, Fi, -1.f);
    // r_n and f16 activation input
    k_rn<<<EW_G, EW_B, 0, stream>>>(xR, xI, t2R, t2I, tR, tI, bR, bI,
                                    lam1, lam2, i, rR, rI, actR);
    // main conv chain
    _Float16* pw = packW;
    k_conv<<<dim3(7200), dim3(128), 0, stream>>>(actR,  pw + (size_t)(0 * NITER + i) * PACKH, actA, 1);
    k_conv<<<dim3(7200), dim3(128), 0, stream>>>(actA,  pw + (size_t)(1 * NITER + i) * PACKH, actB, 1);
    k_conv<<<dim3(7200), dim3(128), 0, stream>>>(actB,  pw + (size_t)(2 * NITER + i) * PACKH, actX3, 0);
    k_soft<<<dim3(28800), dim3(256), 0, stream>>>(actX3, actA, soft_thr, i);
    k_conv<<<dim3(7200), dim3(128), 0, stream>>>(actA,  pw + (size_t)(3 * NITER + i) * PACKH, actB, 1);
    k_conv<<<dim3(7200), dim3(128), 0, stream>>>(actB,  pw + (size_t)(4 * NITER + i) * PACKH, actA, 1);
    k_conv<<<dim3(7200), dim3(128), 0, stream>>>(actA,  pw + (size_t)(5 * NITER + i) * PACKH, actB, 0);
    k_combine<<<EW_G, EW_B, 0, stream>>>(actB, rR, rI, xR, xI);
    // symmetry path: xs = conv6(conv5(conv4(x3))) - r
    k_conv<<<dim3(7200), dim3(128), 0, stream>>>(actX3, pw + (size_t)(3 * NITER + i) * PACKH, actA, 1);
    k_conv<<<dim3(7200), dim3(128), 0, stream>>>(actA,  pw + (size_t)(4 * NITER + i) * PACKH, actB, 1);
    k_conv<<<dim3(7200), dim3(128), 0, stream>>>(actB,  pw + (size_t)(5 * NITER + i) * PACKH, actA, 0);
    k_xsym<<<EW_G, EW_B, 0, stream>>>(actA, rR, rI,
                                      out + (size_t)NTPIX * 2 + (size_t)i * NTPIX * 2);
    // low-rank: SVD via Gram + Jacobi, t and beta update
    k_gram<<<dim3(324), dim3(256), 0, stream>>>(xR, xI, Gr, Gi);
    k_eig<<<dim3(1), dim3(32), 0, stream>>>(Gr, Gi, thres_coef, i, Cr, Ci);
    k_lowrank<<<dim3(100), dim3(256), 0, stream>>>(xR, xI, Cr, Ci, tR, tI, bR, bI, eta, i);
  }
  k_writeout<<<EW_G, EW_B, 0, stream>>>(xR, xI, out);
}
